// RMHA_78176994722494
// MI455X (gfx1250) — matrix operand order repaired and checked by reading, not compiled, not run on hardware
//
#include <hip/hip_runtime.h>
#include <hip/hip_bf16.h>
#include <math.h>

typedef __attribute__((ext_vector_type(16))) __bf16 v16bf;
typedef __attribute__((ext_vector_type(8)))  float  v8f;

namespace {
constexpr int B_ = 4, L_ = 1024, D_ = 1024, H_ = 16, HD_ = 64, MAXR_ = 32, NR_ = 2 * MAXR_ + 1;
constexpr float INV_SCALE = 0.125f;     // 1/sqrt(HD)
constexpr float NEGINF = -1.0e10f;
}

__device__ __forceinline__ unsigned short bfbits(float f) {
  unsigned u = __builtin_bit_cast(unsigned, f);
  return (unsigned short)((u + 0x7FFFu + ((u >> 16) & 1u)) >> 16);   // RNE
}
__device__ __forceinline__ __bf16 f2bf(float f) {
  unsigned short s = bfbits(f);
  return __builtin_bit_cast(__bf16, s);
}
__device__ __forceinline__ float bf2f(__bf16 x) {
  unsigned u = ((unsigned)__builtin_bit_cast(unsigned short, x)) << 16;
  return __builtin_bit_cast(float, u);
}

__device__ __forceinline__ float wave_max(float v) {
  for (int off = 16; off > 0; off >>= 1) v = fmaxf(v, __shfl_xor(v, off, 32));
  return v;
}
__device__ __forceinline__ float wave_sum(float v) {
  for (int off = 16; off > 0; off >>= 1) v += __shfl_xor(v, off, 32);
  return v;
}

// CDNA5 async DMA: global -> LDS, 16B per lane, tracked by ASYNCcnt.
// lds_off is the byte offset inside this workgroup's dynamic LDS allocation.
__device__ __forceinline__ void async_b128(int lds_off, const void* gaddr) {
  asm volatile("global_load_async_to_lds_b128 %0, %1, off"
               :: "v"(lds_off), "v"(gaddr) : "memory");
}
__device__ __forceinline__ void wait_async_le3() {
  asm volatile("s_wait_asynccnt 0x3" ::: "memory");
}
__device__ __forceinline__ void wait_async_0() {
  asm volatile("s_wait_asynccnt 0x0" ::: "memory");
}

#define WMMA_BF16(a, b, c) \
  __builtin_amdgcn_wmma_f32_16x16x32_bf16(false, (a), false, (b), (short)0, (c), false, false)

// ---------------------------------------------------------------------------
// fp32 -> bf16 elementwise (one-time conversion; compute-bound pipeline after)
// ---------------------------------------------------------------------------
__global__ __launch_bounds__(256) void cvt_bf16(const float* __restrict__ in,
                                                unsigned short* __restrict__ out, int n4) {
  int i = blockIdx.x * 256 + threadIdx.x;
  if (i < n4) {
    float4 f = ((const float4*)in)[i];
    ushort4 o;
    o.x = bfbits(f.x); o.y = bfbits(f.y); o.z = bfbits(f.z); o.w = bfbits(f.w);
    ((ushort4*)out)[i] = o;
  }
}

// ---------------------------------------------------------------------------
// C[M,N] = A[M,K] @ W[N,K]^T + bias[N]; A,W bf16; out bf16 or fp32.
// Block tile 64x128, 8 waves (4m x 2n), each wave: 1 A-frag + 4 B-frags ->
// 4 WMMAs on 4 independent acc chains per 32-wide k-step (no hazard padding).
// LDS double-buffered, filled by async B128 DMA (3 per thread per stage).
// ---------------------------------------------------------------------------
template <bool OUT_BF16>
__global__ __launch_bounds__(256) void gemm_bf16(
    const __bf16* __restrict__ A, const __bf16* __restrict__ W,
    const float* __restrict__ bias, void* __restrict__ Cv,
    int M, int N, int K)
{
  extern __shared__ char smem[];   // [2] x { A 64x32 bf16 (4KB) | W 128x32 bf16 (8KB) }
  constexpr int BUF = 12288;
  const int bm = blockIdx.x * 64;
  const int bn = blockIdx.y * 128;
  const int tid = threadIdx.x;
  const int wave = tid >> 5, lane = tid & 31;
  const int wm = (wave & 3) * 16;
  const int wn = (wave >> 2) * 64;

  // per-thread async chunks (16B = 8 bf16 each): 1 of A tile, 2 of W tile
  const int ar = tid >> 2, ac = (tid & 3) << 3;           // A: 64 rows x 4 chunks
  const __bf16* agp  = A + (size_t)(bm + ar) * K + ac;
  const __bf16* wgp0 = W + (size_t)(bn + ar) * K + ac;    // W rows 0..63
  const __bf16* wgp1 = W + (size_t)(bn + 64 + ar) * K + ac; // W rows 64..127
  const int ldsA  = ar * 64 + ac * 2;
  const int ldsW0 = 4096 + ar * 64 + ac * 2;
  const int ldsW1 = 4096 + (64 + ar) * 64 + ac * 2;

  v8f acc[4] = {};
  const int nk = K >> 5;

  async_b128(ldsA, agp);
  async_b128(ldsW0, wgp0);
  async_b128(ldsW1, wgp1);

  for (int t = 0; t < nk; ++t) {
    if (t + 1 < nk) {                     // stage next tile into other buffer
      int base = ((t + 1) & 1) * BUF;
      int k0n = (t + 1) << 5;
      async_b128(base + ldsA, agp + k0n);
      async_b128(base + ldsW0, wgp0 + k0n);
      async_b128(base + ldsW1, wgp1 + k0n);
      wait_async_le3();                   // current tile's 3 DMAs retired (in-order)
    } else {
      wait_async_0();
    }
    __syncthreads();

    const __bf16* As = (const __bf16*)(smem + (t & 1) * BUF);
    const __bf16* Ws = As + 2048;

    // A fragment: 16x32, lane<16 -> K 0..15, lane>=16 -> K 16..31
    v16bf a;
    {
      int m = wm + (lane & 15);
      int kb = (lane >> 4) << 4;
      for (int i = 0; i < 16; ++i)
        a[i] = As[m * 32 + kb + i];
    }
    // 4 B fragments: 32x16, lane = n, lane<16 -> K 0..15, lane>=16 -> K 16..31
    {
      int nb = wn + (lane & 15);
      int kd = (lane >> 4) << 4;
      for (int j = 0; j < 4; ++j) {
        v16bf bf;
        for (int i = 0; i < 16; ++i) bf[i] = Ws[(nb + j * 16) * 32 + kd + i];
        acc[j] = WMMA_BF16(a, bf, acc[j]);
      }
    }
    __syncthreads();
  }

  for (int j = 0; j < 4; ++j)
    for (int r = 0; r < 8; ++r) {
      int m = bm + wm + r + ((lane >> 4) << 3);
      int n = bn + wn + j * 16 + (lane & 15);
      float c = acc[j][r] + bias[n];
      if (OUT_BF16) ((__bf16*)Cv)[(size_t)m * N + n] = f2bf(c);
      else          ((float*)Cv)[(size_t)m * N + n] = c;
    }
}

// ---------------------------------------------------------------------------
// Fused relative-position attention for one (b, h, 32-row q tile).
// q/k/v are bf16; scores fp32 in LDS; double softmax; bf16 x output.
// LDS layout (bytes): Qs[0,4096) | S[4096,135168) | RL | PL | OS
// ---------------------------------------------------------------------------
__global__ __launch_bounds__(256) void rmha_attn(
    const __bf16* __restrict__ qb, const __bf16* __restrict__ kb, const __bf16* __restrict__ vb,
    const float* __restrict__ qm, const float* __restrict__ km,
    const float* __restrict__ relk, const float* __restrict__ relv,
    __bf16* __restrict__ xout)
{
  const int bh = blockIdx.y;
  const int b  = bh >> 4;
  const int h  = bh & (H_ - 1);
  const int q0 = blockIdx.x * 32;
  const int tid = threadIdx.x;
  const int wave = tid >> 5;
  const int lane = tid & 31;

  extern __shared__ char smem[];
  __bf16* Qs = (__bf16*)smem;                          // 32 x 64 bf16
  float*  S  = (float*)(smem + 4096);                  // 32 x 1024 fp32
  float*  RL = (float*)(smem + 4096 + 32 * L_ * 4);    // 32 x 68
  float*  PL = RL + 32 * 68;                           // 32 x 68
  float*  OS = PL + 32 * 68;                           // 32 x 64

  // Async DMA the bf16 Q tile into LDS (1 x B128 per thread, 4KB total)
  {
    int r = tid >> 3, c = (tid & 7) << 3;
    const __bf16* src = qb + ((size_t)b * L_ + q0 + r) * D_ + h * HD_ + c;
    async_b128(r * 128 + c * 2, src);
  }

  // Rel logits: RL[r][rr] = dot(q_row, rel_k[rr])   (overlaps with the DMA)
  for (int i = tid; i < 32 * NR_; i += 256) {
    int r = i / NR_, rr = i % NR_;
    const __bf16* qp = qb + ((size_t)b * L_ + q0 + r) * D_ + h * HD_;
    const float*  rp = relk + rr * HD_;
    float acc = 0.f;
    for (int d = 0; d < HD_; ++d) acc += bf2f(qp[d]) * rp[d];
    RL[r * 68 + rr] = acc;
  }
  wait_async_0();
  __syncthreads();

  // Per-wave A fragments of Q (2 row-halves x 2 K-chunks), reused all k-tiles
  v16bf afr[2][2];
  for (int mh = 0; mh < 2; ++mh)
    for (int kc = 0; kc < 2; ++kc) {
      int m = mh * 16 + (lane & 15);
      int kbse = kc * 32 + ((lane >> 4) << 4);
      v16bf t;
      for (int i = 0; i < 16; ++i)
        t[i] = Qs[m * HD_ + kbse + i];
      afr[mh][kc] = t;
    }

  // S = Q K^T : each wave processes TWO 16-column tiles per iteration ->
  // 8 WMMAs on 4 independent acc chains (hides WMMA->WMMA RAW latency).
  // B frags are direct 32B-aligned v16bf global loads (zero conversion VALU).
  for (int kt = wave; kt < L_ / 16; kt += 16) {
    int nA = kt * 16 + (lane & 15);
    int nB = (kt + 8) * 16 + (lane & 15);
    v16bf bA[2], bB[2];
    for (int kc = 0; kc < 2; ++kc) {
      int koff = h * HD_ + kc * 32 + ((lane >> 4) << 4);
      bA[kc] = *(const v16bf*)(kb + ((size_t)b * L_ + nA) * D_ + koff);
      bB[kc] = *(const v16bf*)(kb + ((size_t)b * L_ + nB) * D_ + koff);
    }
    v8f acc[4] = {};                      // [tileA mh0, tileA mh1, tileB mh0, tileB mh1]
    for (int kc = 0; kc < 2; ++kc) {
      acc[0] = WMMA_BF16(afr[0][kc], bA[kc], acc[0]);
      acc[1] = WMMA_BF16(afr[1][kc], bA[kc], acc[1]);
      acc[2] = WMMA_BF16(afr[0][kc], bB[kc], acc[2]);
      acc[3] = WMMA_BF16(afr[1][kc], bB[kc], acc[3]);
    }
    for (int tmh = 0; tmh < 4; ++tmh) {
      int n = (tmh & 2) ? nB : nA;
      int mh = tmh & 1;
      for (int r = 0; r < 8; ++r) {
        int m = mh * 16 + r + ((lane >> 4) << 3);
        S[m * L_ + n] = acc[tmh][r];
      }
    }
  }
  __syncthreads();

  // scale + rel term + mask
  for (int i = tid; i < 32 * L_; i += 256) {
    int r = i >> 10, c = i & (L_ - 1);
    int qi = q0 + r;
    int d = c - qi;
    d = d < -MAXR_ ? -MAXR_ : (d > MAXR_ ? MAXR_ : d);
    float s = (S[i] + RL[r * 68 + d + MAXR_]) * INV_SCALE;
    bool msk = (qm[b * L_ + qi] * km[b * L_ + c]) != 0.f;
    S[i] = s + (msk ? 0.f : NEGINF);
  }
  __syncthreads();

  // Double softmax per row (reference semantics), 4 rows per wave
  for (int rr = 0; rr < 4; ++rr) {
    int r = wave * 4 + rr;
    float* row = S + r * L_;
    float qmv = qm[b * L_ + q0 + r];
    float mx = -3.0e38f;
    for (int c = lane; c < L_; c += 32) mx = fmaxf(mx, row[c]);
    mx = wave_max(mx);
    float sm = 0.f;
    for (int c = lane; c < L_; c += 32) { float e = __expf(row[c] - mx); row[c] = e; sm += e; }
    sm = wave_sum(sm);
    float inv = 1.f / sm;
    for (int c = lane; c < L_; c += 32) {
      float p = row[c] * inv;
      bool msk = (qmv * km[b * L_ + c]) != 0.f;
      row[c] = msk ? p : 0.f;
    }
    float mx2 = -3.0e38f;
    for (int c = lane; c < L_; c += 32) mx2 = fmaxf(mx2, row[c]);
    mx2 = wave_max(mx2);
    float sm2 = 0.f;
    for (int c = lane; c < L_; c += 32) { float e = __expf(row[c] - mx2); row[c] = e; sm2 += e; }
    sm2 = wave_sum(sm2);
    float inv2 = 1.f / sm2;
    for (int c = lane; c < L_; c += 32) row[c] *= inv2;
  }
  __syncthreads();

  // Pooled probabilities over clipped relative buckets (one thread per q row)
  if (tid < 32) {
    float* pl = PL + tid * 68;
    for (int rr = 0; rr < NR_; ++rr) pl[rr] = 0.f;
    const float* row = S + tid * L_;
    int qi = q0 + tid;
    for (int c = 0; c < L_; ++c) {
      int d = c - qi;
      d = d < -MAXR_ ? -MAXR_ : (d > MAXR_ ? MAXR_ : d);
      pl[d + MAXR_] += row[c];
    }
  }

  // O = P @ V : each wave owns one 16x16 output fragment
  {
    int mh = wave & 1;
    int dq = wave >> 1;
    int n  = dq * 16 + (lane & 15);
    v8f acc = {};
    for (int k0 = 0; k0 < L_; k0 += 32) {
      v16bf a, bf;
      int m = mh * 16 + (lane & 15);
      int kbse = (lane >> 4) << 4;
      for (int i = 0; i < 16; ++i)
        a[i] = f2bf(S[m * L_ + k0 + kbse + i]);
      int kd0 = (lane >> 4) << 4;
      for (int i = 0; i < 16; ++i)
        bf[i] = vb[((size_t)b * L_ + k0 + kd0 + i) * D_ + h * HD_ + n];
      acc = WMMA_BF16(a, bf, acc);
    }
    for (int r = 0; r < 8; ++r) {
      int m = mh * 16 + r + ((lane >> 4) << 3);
      OS[m * HD_ + n] = acc[r];
    }
  }
  __syncthreads();

  // Add pooled @ rel_v and store bf16 per-head output slice (b, l, D) layout
  for (int i = tid; i < 32 * HD_; i += 256) {
    int r = i >> 6, d = i & 63;
    float acc = OS[i];
    const float* pl = PL + r * 68;
    for (int rr = 0; rr < NR_; ++rr) acc += pl[rr] * relv[rr * HD_ + d];
    xout[((size_t)b * L_ + q0 + r) * D_ + h * HD_ + d] = f2bf(acc);
  }
}

extern "C" void kernel_launch(void* const* d_in, const int* in_sizes, int n_in,
                              void* d_out, int out_size, void* d_ws, size_t ws_size,
                              hipStream_t stream) {
  const float* query = (const float*)d_in[0];
  const float* key   = (const float*)d_in[1];
  const float* value = (const float*)d_in[2];
  const float* qmask = (const float*)d_in[3];
  const float* kmask = (const float*)d_in[4];
  const float* WQw = (const float*)d_in[5];
  const float* WQb = (const float*)d_in[6];
  const float* WKw = (const float*)d_in[7];
  const float* WKb = (const float*)d_in[8];
  const float* WVw = (const float*)d_in[9];
  const float* WVb = (const float*)d_in[10];
  const float* WOw = (const float*)d_in[11];
  const float* WOb = (const float*)d_in[12];
  const float* relk = (const float*)d_in[13];
  const float* relv = (const float*)d_in[14];
  float* out = (float*)d_out;

  const size_t nBLD = (size_t)B_ * L_ * D_;   // 4M
  const size_t nDD  = (size_t)D_ * D_;        // 1M
  __bf16* w   = (__bf16*)d_ws;
  __bf16* qI  = w;                 // bf16 copies of inputs
  __bf16* kI  = qI + nBLD;
  __bf16* vI  = kI + nBLD;
  __bf16* wqB = vI + nBLD;
  __bf16* wkB = wqB + nDD;
  __bf16* wvB = wkB + nDD;
  __bf16* woB = wvB + nDD;
  __bf16* qP  = woB + nDD;         // projected q/k/v and attention output
  __bf16* kP  = qP + nBLD;
  __bf16* vP  = kP + nBLD;
  __bf16* xB  = vP + nBLD;

  // one-time fp32 -> bf16 conversions (bandwidth-trivial at 23.3 TB/s)
  auto cvt = [&](const float* src, __bf16* dst, size_t n) {
    int n4 = (int)(n >> 2);
    cvt_bf16<<<(n4 + 255) / 256, 256, 0, stream>>>(src, (unsigned short*)dst, n4);
  };
  cvt(query, qI, nBLD); cvt(key, kI, nBLD); cvt(value, vI, nBLD);
  cvt(WQw, wqB, nDD);   cvt(WKw, wkB, nDD); cvt(WVw, wvB, nDD); cvt(WOw, woB, nDD);

  dim3 ggrid(B_ * L_ / 64, D_ / 128);
  size_t gsm = 2 * 12288;                     // double-buffered 2 x (4KB A + 8KB W)
  gemm_bf16<true><<<ggrid, 256, gsm, stream>>>(qI, wqB, WQb, qP, B_ * L_, D_, D_);
  gemm_bf16<true><<<ggrid, 256, gsm, stream>>>(kI, wkB, WKb, kP, B_ * L_, D_, D_);
  gemm_bf16<true><<<ggrid, 256, gsm, stream>>>(vI, wvB, WVb, vP, B_ * L_, D_, D_);

  size_t asm_sz = (size_t)4096                // Qs (bf16)
                + (size_t)32 * L_ * 4         // S
                + (size_t)32 * 68 * 4 * 2     // RL + PL
                + (size_t)32 * 64 * 4;        // OS
  hipFuncSetAttribute(reinterpret_cast<const void*>(rmha_attn),
                      hipFuncAttributeMaxDynamicSharedMemorySize, (int)asm_sz);
  rmha_attn<<<dim3(L_ / 32, B_ * H_), 256, asm_sz, stream>>>(
      qP, kP, vP, qmask, kmask, relk, relv, xB);

  gemm_bf16<false><<<ggrid, 256, gsm, stream>>>(xB, woB, WOb, out, B_ * L_, D_, D_);
}
